// AGGT_51436528337258
// MI455X (gfx1250) — matrix operand order repaired and checked by reading, not compiled, not run on hardware
//
#include <hip/hip_runtime.h>
#include <hip/hip_bf16.h>

#define NNODES 100000
#define NEDGES 1600000
#define DIN    128
#define H      64
#define DOUT   64
#define NTILES (NNODES / 16)   // 6250, exact

typedef __attribute__((ext_vector_type(16))) __bf16        v16bf;
typedef __attribute__((ext_vector_type(16))) unsigned short us16;
typedef __attribute__((ext_vector_type(8)))  float          v8f;

union BFCast { us16 u; v16bf b; };

// ---- WMMA fragment loaders (CDNA5 V_WMMA_F32_16X16X32_BF16 layouts) ----

// A: 16x32 bf16, row-major f32 source (LDS tile), per-lane M=lane&15,
// K = 2*(j&3)+p + 8*(lane>=16) + 16*(j>=4)  with e=2j+p.
// Native (__bf16) casts so clang emits packed v_cvt bf16 ops.
static __device__ __forceinline__ v16bf load_a_frag(const float* src, int ld) {
  int lane = threadIdx.x & 31;
  int m = lane & 15, hh = lane >> 4;
  v16bf out;
#pragma unroll
  for (int e = 0; e < 16; ++e) {
    int j = e >> 1, p = e & 1;
    int k = ((j & 3) << 1) + p + (hh << 3) + ((j >> 2) << 4);
    out[e] = (__bf16)src[m * ld + k];
  }
  return out;
}

// B: 32x16 bf16 — lane L holds column N=L&15, K = (e&7) + 8*(L>=16) + 16*(e>=8).
// Row-major bf16 weight buffer, strided loads.
static __device__ __forceinline__ v16bf load_b_frag(const unsigned short* w, int ldb) {
  int lane = threadIdx.x & 31;
  int n = lane & 15, hh = lane >> 4;
  BFCast cv;
#pragma unroll
  for (int e = 0; e < 16; ++e) {
    int k = (e & 7) + (hh << 3) + ((e >> 3) << 4);
    cv.u[e] = w[(size_t)k * ldb + n];
  }
  return cv.b;
}

#define WMMA_BF16(a, b, c) \
  __builtin_amdgcn_wmma_f32_16x16x32_bf16(false, (a), false, (b), (short)0, (c), false, false)

// ---------------------------------------------------------------------------
__global__ void k_convert_bf16(const float* __restrict__ src,
                               unsigned short* __restrict__ dst, int n) {
  int i = blockIdx.x * blockDim.x + threadIdx.x;
  if (i < n) {
    union { __bf16 b; unsigned short u; } cv;
    cv.b = (__bf16)src[i];
    dst[i] = cv.u;
  }
}

// K1: h = relu(LN(x@Wt + bt)) -> cat[:,0:64] ; g = relu(x@Wg + bg) -> cat[:,64:128]
__global__ void __launch_bounds__(128)
k_input_layer(const float* __restrict__ x,
              const unsigned short* __restrict__ wT, const unsigned short* __restrict__ wG,
              const float* __restrict__ tb, const float* __restrict__ g0w,
              const float* __restrict__ b0w, const float* __restrict__ gb,
              float* __restrict__ cat) {
  __shared__ float xs[16][DIN + 4];
  __shared__ float ts[16][H + 2];
  __shared__ float mu_s[16], rs_s[16];
  int tid = threadIdx.x;
  int row0 = blockIdx.x * 16;
  for (int idx = tid; idx < 16 * DIN; idx += 128) {
    int r = idx >> 7, c = idx & 127;
    xs[r][c] = x[(size_t)(row0 + r) * DIN + c];
  }
  __syncthreads();
  int wave = tid >> 5, lane = tid & 31;
  v8f accT = {}; v8f accG = {};
#pragma unroll
  for (int kt = 0; kt < DIN / 32; ++kt) {
    v16bf a  = load_a_frag(&xs[0][kt * 32], DIN + 4);
    v16bf bT = load_b_frag(wT + (size_t)(kt * 32) * H + wave * 16, H);
    v16bf bG = load_b_frag(wG + (size_t)(kt * 32) * H + wave * 16, H);
    accT = WMMA_BF16(a, bT, accT);
    accG = WMMA_BF16(a, bG, accG);
  }
  int col = wave * 16 + (lane & 15);
  int rbase = (lane >> 4) * 8;
  float tbias = tb[col], gbias = gb[col];
  float tval[8];
#pragma unroll
  for (int r = 0; r < 8; ++r) {
    int row = rbase + r;
    tval[r] = accT[r] + tbias;
    ts[row][col] = tval[r];
    float gv = accG[r] + gbias;
    cat[(size_t)(row0 + row) * 128 + 64 + col] = gv > 0.f ? gv : 0.f;
  }
  __syncthreads();
  if (tid < 16) {
    float s = 0.f, ss = 0.f;
#pragma unroll
    for (int j = 0; j < H; ++j) { float vv = ts[tid][j]; s += vv; ss += vv * vv; }
    float mu = s * (1.f / H);
    float var = ss * (1.f / H) - mu * mu;
    mu_s[tid] = mu;
    rs_s[tid] = rsqrtf(var + 1e-5f);
  }
  __syncthreads();
  float lg = g0w[col], lb = b0w[col];
#pragma unroll
  for (int r = 0; r < 8; ++r) {
    int row = rbase + r;
    float hv = (tval[r] - mu_s[row]) * rs_s[row] * lg + lb;
    cat[(size_t)(row0 + row) * 128 + col] = hv > 0.f ? hv : 0.f;
  }
}

// K2: q,k,v = h @ {Wq,Wk,Wv} + bias
__global__ void __launch_bounds__(128)
k_qkv(const float* __restrict__ cat,
      const unsigned short* __restrict__ wq, const unsigned short* __restrict__ wk,
      const unsigned short* __restrict__ wv,
      const float* __restrict__ qb, const float* __restrict__ kb, const float* __restrict__ vb,
      float* __restrict__ q, float* __restrict__ k, float* __restrict__ v) {
  __shared__ float hs[16][H + 2];
  int tid = threadIdx.x;
  int row0 = blockIdx.x * 16;
  for (int idx = tid; idx < 16 * H; idx += 128) {
    int r = idx >> 6, c = idx & 63;
    hs[r][c] = cat[(size_t)(row0 + r) * 128 + c];
  }
  __syncthreads();
  int wave = tid >> 5, lane = tid & 31;
  v8f aq = {}, ak = {}, av = {};
#pragma unroll
  for (int kt = 0; kt < H / 32; ++kt) {
    v16bf a = load_a_frag(&hs[0][kt * 32], H + 2);
    aq = WMMA_BF16(a, load_b_frag(wq + (size_t)(kt * 32) * H + wave * 16, H), aq);
    ak = WMMA_BF16(a, load_b_frag(wk + (size_t)(kt * 32) * H + wave * 16, H), ak);
    av = WMMA_BF16(a, load_b_frag(wv + (size_t)(kt * 32) * H + wave * 16, H), av);
  }
  int col = wave * 16 + (lane & 15);
  int rbase = (lane >> 4) * 8;
  float qbi = qb[col], kbi = kb[col], vbi = vb[col];
#pragma unroll
  for (int r = 0; r < 8; ++r) {
    size_t o = (size_t)(row0 + rbase + r) * H + col;
    q[o] = aq[r] + qbi;
    k[o] = ak[r] + kbi;
    v[o] = av[r] + vbi;
  }
}

// K3: qss = sum q^2, kss = sum k^2, ksum[64] = column sums of k
__global__ void __launch_bounds__(256)
k_reduce(const float* __restrict__ q, const float* __restrict__ k,
         float* __restrict__ ksum, float* __restrict__ scal) {
  __shared__ float s_col[64];
  __shared__ float s_q, s_k;
  int tid = threadIdx.x;
  if (tid < 64) s_col[tid] = 0.f;
  if (tid == 0) { s_q = 0.f; s_k = 0.f; }
  __syncthreads();
  float lq = 0.f, lk = 0.f;
  size_t total = (size_t)NNODES * H;
  for (size_t i = (size_t)blockIdx.x * blockDim.x + tid; i < total;
       i += (size_t)gridDim.x * blockDim.x) {
    float qv = q[i]; lq += qv * qv;
    float kv = k[i]; lk += kv * kv;
    atomicAdd(&s_col[(int)(i & 63)], kv);
  }
  atomicAdd(&s_q, lq);
  atomicAdd(&s_k, lk);
  __syncthreads();
  if (tid < 64) atomicAdd(&ksum[tid], s_col[tid]);
  if (tid == 0) { atomicAdd(&scal[0], s_q); atomicAdd(&scal[1], s_k); }
}

// K4: kvs_raw[m][d] = sum_n k[n][m] * v[n][d]   (K = 100000 via 32-node chunks)
__global__ void __launch_bounds__(128)
k_kvs(const float* __restrict__ k, const float* __restrict__ v, float* __restrict__ kvs) {
  __shared__ float ks_s[32][H + 2];
  __shared__ float vs_s[32][H + 2];
  int tid = threadIdx.x;
  int wave = tid >> 5, lane = tid & 31;
  int m = lane & 15, hh = lane >> 4;
  v8f acc0 = {}, acc1 = {}, acc2 = {}, acc3 = {};
  int nchunks = NNODES / 32;
  for (int c = blockIdx.x; c < nchunks; c += gridDim.x) {
    size_t base = (size_t)c * 32 * H;
    for (int idx = tid; idx < 32 * H; idx += 128) {
      int r = idx >> 6, cc = idx & 63;
      ks_s[r][cc] = k[base + (size_t)r * H + cc];
      vs_s[r][cc] = v[base + (size_t)r * H + cc];
    }
    __syncthreads();
    // A = k_chunk^T: A[mrow][kk] = ks_s[kk][wave*16+mrow]
    v16bf a;
#pragma unroll
    for (int e = 0; e < 16; ++e) {
      int j = e >> 1, p = e & 1;
      int kk = ((j & 3) << 1) + p + (hh << 3) + ((j >> 2) << 4);
      a[e] = (__bf16)ks_s[kk][wave * 16 + m];
    }
#pragma unroll
    for (int dt = 0; dt < 4; ++dt) {
      v16bf b;
#pragma unroll
      for (int e = 0; e < 16; ++e) b[e] = (__bf16)vs_s[(e & 7) + (hh << 3) + ((e >> 3) << 4)][dt * 16 + m];
      v8f* accp = (dt == 0) ? &acc0 : (dt == 1) ? &acc1 : (dt == 2) ? &acc2 : &acc3;
      *accp = WMMA_BF16(a, b, *accp);
    }
    __syncthreads();
  }
  int rbase = hh * 8;
  v8f accs[4] = {acc0, acc1, acc2, acc3};
#pragma unroll
  for (int dt = 0; dt < 4; ++dt)
#pragma unroll
    for (int r = 0; r < 8; ++r)
      atomicAdd(&kvs[(size_t)(wave * 16 + rbase + r) * H + dt * 16 + m], accs[dt][r]);
}

// K5: num = (q@kvs_raw)/(|q||k|) + N*v ; denom = (q.ksum)/(|q||k|) + N ;
//     x_trans = relu(LN((num/denom + h0)/2))  -> cat[:,0:64]
__global__ void __launch_bounds__(128)
k_attn_out(const float* __restrict__ q, const float* __restrict__ v,
           const unsigned short* __restrict__ kvsbf,
           const float* __restrict__ ksum, const float* __restrict__ scal,
           const float* __restrict__ g1, const float* __restrict__ b1,
           float* __restrict__ cat) {
  __shared__ float qs_s[16][H + 2];
  __shared__ float ts[16][H + 2];
  __shared__ float ksum_s[64];
  __shared__ float denom_s[16];
  __shared__ float mu_s[16], rs_s[16];
  int tid = threadIdx.x;
  int row0 = blockIdx.x * 16;
  for (int idx = tid; idx < 16 * H; idx += 128) {
    int r = idx >> 6, c = idx & 63;
    qs_s[r][c] = q[(size_t)(row0 + r) * H + c];
  }
  if (tid < 64) ksum_s[tid] = ksum[tid];
  __syncthreads();
  float invqk = rsqrtf(scal[0]) * rsqrtf(scal[1]);
  const float fN = (float)NNODES;
  if (tid < 16) {
    float d = 0.f;
#pragma unroll
    for (int j = 0; j < H; ++j) d += qs_s[tid][j] * ksum_s[j];
    denom_s[tid] = d * invqk + fN;
  }
  int wave = tid >> 5, lane = tid & 31;
  v8f acc = {};
#pragma unroll
  for (int kt = 0; kt < H / 32; ++kt) {
    v16bf a = load_a_frag(&qs_s[0][kt * 32], H + 2);
    v16bf b = load_b_frag(kvsbf + (size_t)(kt * 32) * H + wave * 16, H);
    acc = WMMA_BF16(a, b, acc);
  }
  __syncthreads();
  int col = wave * 16 + (lane & 15);
  int rbase = (lane >> 4) * 8;
  float pre[8];
#pragma unroll
  for (int r = 0; r < 8; ++r) {
    int row = rbase + r;
    size_t o = (size_t)(row0 + row) * H + col;
    float num = acc[r] * invqk + fN * v[o];
    float hattn = num / denom_s[row];
    float h0 = cat[(size_t)(row0 + row) * 128 + col];
    pre[r] = 0.5f * (hattn + h0);
    ts[row][col] = pre[r];
  }
  __syncthreads();
  if (tid < 16) {
    float s = 0.f, ss = 0.f;
#pragma unroll
    for (int j = 0; j < H; ++j) { float xv = ts[tid][j]; s += xv; ss += xv * xv; }
    float mu = s * (1.f / H);
    float var = ss * (1.f / H) - mu * mu;
    mu_s[tid] = mu;
    rs_s[tid] = rsqrtf(var + 1e-5f);
  }
  __syncthreads();
  float lg = g1[col], lb = b1[col];
#pragma unroll
  for (int r = 0; r < 8; ++r) {
    int row = rbase + r;
    float xv = (pre[r] - mu_s[row]) * rs_s[row] * lg + lb;
    cat[(size_t)(row0 + row) * 128 + col] = xv > 0.f ? xv : 0.f;
  }
}

// K6: in-degree on col
__global__ void __launch_bounds__(256)
k_deg(const int* __restrict__ ei, float* __restrict__ deg) {
  for (int e = blockIdx.x * blockDim.x + threadIdx.x; e < NEDGES;
       e += gridDim.x * blockDim.x)
    atomicAdd(&deg[ei[NEDGES + e]], 1.0f);
}

// K7: agg[col] += dinv[col]*dinv[row] * g[row]  (one wave per edge, lanes = features)
__global__ void __launch_bounds__(256)
k_spmm(const int* __restrict__ ei, const float* __restrict__ deg,
       const float* __restrict__ cat, float* __restrict__ agg) {
  int lane = threadIdx.x & 31;
  int wid = (blockIdx.x * blockDim.x + threadIdx.x) >> 5;
  int nw = (gridDim.x * blockDim.x) >> 5;
  for (int e = wid; e < NEDGES; e += nw) {
    // prefetch next iteration's edge-index words (global_prefetch_b8)
    if (e + nw < NEDGES) {
      __builtin_prefetch(&ei[e + nw], 0, 3);
      __builtin_prefetch(&ei[NEDGES + e + nw], 0, 3);
    }
    int row = ei[e];
    int col = ei[NEDGES + e];
    float dr = deg[row], dc = deg[col];
    float val = (dr > 0.f ? rsqrtf(dr) : 0.f) * (dc > 0.f ? rsqrtf(dc) : 0.f);
    const float* gr = cat + (size_t)row * 128 + 64;
    float* ao = agg + (size_t)col * 64;
    atomicAdd(&ao[lane], val * gr[lane]);
    atomicAdd(&ao[lane + 32], val * gr[lane + 32]);
  }
}

// K8: x_gnn = relu(agg@gc + b) + g0  -> cat[:,64:128]
__global__ void __launch_bounds__(128)
k_gnn_out(const float* __restrict__ agg, const unsigned short* __restrict__ wgc,
          const float* __restrict__ gcb, float* __restrict__ cat) {
  __shared__ float as[16][H + 2];
  int tid = threadIdx.x;
  int row0 = blockIdx.x * 16;
  for (int idx = tid; idx < 16 * H; idx += 128) {
    int r = idx >> 6, c = idx & 63;
    as[r][c] = agg[(size_t)(row0 + r) * H + c];
  }
  __syncthreads();
  int wave = tid >> 5, lane = tid & 31;
  v8f acc = {};
#pragma unroll
  for (int kt = 0; kt < H / 32; ++kt) {
    v16bf a = load_a_frag(&as[0][kt * 32], H + 2);
    v16bf b = load_b_frag(wgc + (size_t)(kt * 32) * H + wave * 16, H);
    acc = WMMA_BF16(a, b, acc);
  }
  int col = wave * 16 + (lane & 15);
  int rbase = (lane >> 4) * 8;
  float bi = gcb[col];
#pragma unroll
  for (int r = 0; r < 8; ++r) {
    size_t o = (size_t)(row0 + rbase + r) * 128 + 64 + col;
    float xv = acc[r] + bi;
    xv = xv > 0.f ? xv : 0.f;
    cat[o] = xv + cat[o];
  }
}

// K9: out = cat @ fc_out_w + fc_out_b
__global__ void __launch_bounds__(128)
k_final(const float* __restrict__ cat, const unsigned short* __restrict__ wout,
        const float* __restrict__ ob, float* __restrict__ out) {
  __shared__ float cs[16][128 + 4];
  int tid = threadIdx.x;
  int row0 = blockIdx.x * 16;
  for (int idx = tid; idx < 16 * 128; idx += 128) {
    int r = idx >> 7, c = idx & 127;
    cs[r][c] = cat[(size_t)(row0 + r) * 128 + c];
  }
  __syncthreads();
  int wave = tid >> 5, lane = tid & 31;
  v8f acc = {};
#pragma unroll
  for (int kt = 0; kt < 4; ++kt) {
    v16bf a = load_a_frag(&cs[0][kt * 32], 128 + 4);
    v16bf b = load_b_frag(wout + (size_t)(kt * 32) * DOUT + wave * 16, DOUT);
    acc = WMMA_BF16(a, b, acc);
  }
  int col = wave * 16 + (lane & 15);
  int rbase = (lane >> 4) * 8;
  float bi = ob[col];
#pragma unroll
  for (int r = 0; r < 8; ++r)
    out[(size_t)(row0 + rbase + r) * DOUT + col] = acc[r] + bi;
}

// ---------------------------------------------------------------------------
extern "C" void kernel_launch(void* const* d_in, const int* in_sizes, int n_in,
                              void* d_out, int out_size, void* d_ws, size_t ws_size,
                              hipStream_t stream) {
  (void)in_sizes; (void)n_in; (void)out_size; (void)ws_size;
  const float* x        = (const float*)d_in[0];
  const int*   ei       = (const int*)d_in[1];
  const float* fc_t_w   = (const float*)d_in[2];
  const float* fc_t_b   = (const float*)d_in[3];
  const float* ln0_g    = (const float*)d_in[4];
  const float* ln0_b    = (const float*)d_in[5];
  const float* Wq_w     = (const float*)d_in[6];
  const float* Wq_b     = (const float*)d_in[7];
  const float* Wk_w     = (const float*)d_in[8];
  const float* Wk_b     = (const float*)d_in[9];
  const float* Wv_w     = (const float*)d_in[10];
  const float* Wv_b     = (const float*)d_in[11];
  const float* ln1_g    = (const float*)d_in[12];
  const float* ln1_b    = (const float*)d_in[13];
  const float* fc_g_w   = (const float*)d_in[14];
  const float* fc_g_b   = (const float*)d_in[15];
  const float* gc_w     = (const float*)d_in[16];
  const float* gc_b     = (const float*)d_in[17];
  const float* fc_out_w = (const float*)d_in[18];
  const float* fc_out_b = (const float*)d_in[19];
  float* out = (float*)d_out;

  char* ws = (char*)d_ws;
  const size_t NH4 = (size_t)NNODES * H * 4;
  size_t off = 0;
  auto al = [](size_t o) { return (o + 255) & ~(size_t)255; };
  float* q   = (float*)(ws + off); off = al(off + NH4);
  float* k   = (float*)(ws + off); off = al(off + NH4);
  float* v   = (float*)(ws + off); off = al(off + NH4);
  float* cat = (float*)(ws + off); off = al(off + (size_t)NNODES * 128 * 4);
  size_t zero_off = off;
  float* agg  = (float*)(ws + off); off = al(off + NH4);
  float* deg  = (float*)(ws + off); off = al(off + (size_t)NNODES * 4);
  float* kvs  = (float*)(ws + off); off = al(off + 4096 * 4);
  float* ksum = (float*)(ws + off); off = al(off + 64 * 4);
  float* scal = (float*)(ws + off); off = al(off + 256);
  size_t zero_bytes = off - zero_off;
  unsigned short* wT_bf  = (unsigned short*)(ws + off); off = al(off + 8192 * 2);
  unsigned short* wG_bf  = (unsigned short*)(ws + off); off = al(off + 8192 * 2);
  unsigned short* wq_bf  = (unsigned short*)(ws + off); off = al(off + 4096 * 2);
  unsigned short* wk_bf  = (unsigned short*)(ws + off); off = al(off + 4096 * 2);
  unsigned short* wv_bf  = (unsigned short*)(ws + off); off = al(off + 4096 * 2);
  unsigned short* wgc_bf = (unsigned short*)(ws + off); off = al(off + 4096 * 2);
  unsigned short* wo_bf  = (unsigned short*)(ws + off); off = al(off + 8192 * 2);
  unsigned short* kvs_bf = (unsigned short*)(ws + off); off = al(off + 4096 * 2);

  hipMemsetAsync(ws + zero_off, 0, zero_bytes, stream);

  k_convert_bf16<<<32, 256, 0, stream>>>(fc_t_w, wT_bf, 8192);
  k_convert_bf16<<<32, 256, 0, stream>>>(fc_g_w, wG_bf, 8192);
  k_convert_bf16<<<16, 256, 0, stream>>>(Wq_w, wq_bf, 4096);
  k_convert_bf16<<<16, 256, 0, stream>>>(Wk_w, wk_bf, 4096);
  k_convert_bf16<<<16, 256, 0, stream>>>(Wv_w, wv_bf, 4096);
  k_convert_bf16<<<16, 256, 0, stream>>>(gc_w, wgc_bf, 4096);
  k_convert_bf16<<<32, 256, 0, stream>>>(fc_out_w, wo_bf, 8192);

  k_input_layer<<<NTILES, 128, 0, stream>>>(x, wT_bf, wG_bf, fc_t_b, ln0_g, ln0_b,
                                            fc_g_b, cat);
  k_qkv<<<NTILES, 128, 0, stream>>>(cat, wq_bf, wk_bf, wv_bf, Wq_b, Wk_b, Wv_b, q, k, v);
  k_reduce<<<512, 256, 0, stream>>>(q, k, ksum, scal);
  k_kvs<<<1024, 128, 0, stream>>>(k, v, kvs);
  k_convert_bf16<<<16, 256, 0, stream>>>(kvs, kvs_bf, 4096);
  k_attn_out<<<NTILES, 128, 0, stream>>>(q, v, kvs_bf, ksum, scal, ln1_g, ln1_b, cat);

  k_deg<<<2048, 256, 0, stream>>>(ei, deg);
  k_spmm<<<4096, 256, 0, stream>>>(ei, deg, cat, agg);
  k_gnn_out<<<NTILES, 128, 0, stream>>>(agg, wgc_bf, gc_b, cat);

  k_final<<<NTILES, 128, 0, stream>>>(cat, wo_bf, fc_out_b, out);
}
